// Hier_LCN_71880572666226
// MI455X (gfx1250) — matrix operand order repaired and checked by reading, not compiled, not run on hardware
//
#include <hip/hip_runtime.h>

#define NN 512
#define BB 256
#define CC 64
#define HBATCH 128
#define EPSF 1e-5f

typedef __attribute__((ext_vector_type(16))) __bf16 v16bf;
typedef __attribute__((ext_vector_type(8)))  float  v8f;
typedef __attribute__((ext_vector_type(2)))  __bf16 bf16x2;
typedef __attribute__((ext_vector_type(2)))  float  f32x2;

union BF16Op { unsigned int u[8]; uint4 q[2]; v16bf v; };

// Packed f32x2 -> bf16x2 (RNE); lowers to v_cvt_pk_bf16_f32 on gfx1250.
__device__ __forceinline__ unsigned int pack_bf2(float lo, float hi) {
  f32x2 v; v[0] = lo; v[1] = hi;
  union { bf16x2 h; unsigned int u; } cv;
  cv.h = __builtin_convertvector(v, bf16x2);
  return cv.u;
}
__device__ __forceinline__ unsigned short f2bf(float f) {
  union { __bf16 h; unsigned short u; } cv;
  cv.h = (__bf16)f;
  return cv.u;
}

__device__ __forceinline__ const float* hop_wptr(int m, int n,
    const float* W_S, const float* W_L3, const float* W_L4) {
  if (m < 5) return W_S  + (size_t)(n * 5 + m)       * 64 * 64;
  if (m < 7) return W_L3 + (size_t)(n * 2 + (m - 5)) * 64 * 32;
  return            W_L4 + (size_t)(n * 2 + (m - 7)) * 64 * 64;
}
__device__ __forceinline__ int hop_j(int m, int n,
    const int* idxS, const int* idxL3, const int* idxL4) {
  if (m < 5) return idxS [n * 5 + m];
  if (m < 7) return idxL3[n * 2 + (m - 5)];
  return            idxL4[n * 2 + (m - 7)];
}

// A-operand (16x32 bf16) per ISA 7.12.2: lanes 0-15 hold K=0..7,16..23 ; lanes 16-31 hold K=8..15,24..31
__device__ __forceinline__ v16bf ld_a(const unsigned int* rowbase, int kp0, int hs) {
  BF16Op t;
  t.q[0] = *(const uint4*)(rowbase + kp0 + hs * 4);
  t.q[1] = *(const uint4*)(rowbase + kp0 + 8 + hs * 4);
  return t.v;
}
// B-operand (32x16 bf16), same K order as A: lanes 0-15 hold K=0..7,16..23 ; lanes 16-31 hold K=8..15,24..31
__device__ __forceinline__ v16bf ld_b(const unsigned int* colbase, int kp0, int hs) {
  BF16Op t;
  const unsigned int* p = colbase + kp0 + hs * 4;
  t.q[0] = *(const uint4*)(p);
  t.q[1] = *(const uint4*)(p + 8);
  return t.v;
}

template <int MB, int ME, int D, int R0>
__device__ __forceinline__ void process_group(
    int n, int half, int tid, int w, int hs, int lr,
    const float* __restrict__ feat, const float* __restrict__ Amat,
    const int* __restrict__ idxS, const int* __restrict__ idxL3, const int* __restrict__ idxL4,
    const float* __restrict__ W_S, const float* __restrict__ W_L3, const float* __restrict__ W_L4,
    unsigned int* x_u32, unsigned int* y_u32, unsigned int* wT_u32, const unsigned int* waT_u32,
    v8f (&Hacc)[4])
{
  v8f Yacc[4] = {};

  #pragma unroll
  for (int m = MB; m < ME; ++m) {
    __syncthreads();  // protect x_u32 / wT_u32 from previous readers
    const int j = hop_j(m, n, idxS, idxL3, idxL4);
    const float* Wp = hop_wptr(m, n, W_S, W_L3, W_L4);

    // ---- Software-pipelined staging: issue ALL global loads first, then pack + ds_store. ----
    float  av[8];
    float4 xf[8];
    #pragma unroll
    for (int k = 0; k < 8; ++k) {
      int p = k * 256 + tid;           // 128 rows * 16 quads
      int row = p >> 4, cq = p & 15;
      int b = half * HBATCH + row;
      av[k] = Amat[((size_t)b * NN + n) * NN + j];
      xf[k] = *(const float4*)(feat + ((size_t)b * NN + j) * CC + 4 * cq);
    }
    const int per = D / 8;
    float wlo[8], whi[8];
    {
      int col = tid % D, g = tid / D;
      #pragma unroll
      for (int i = 0; i < per; ++i) {
        int kp = g * per + i;
        wlo[i] = Wp[(2 * kp) * D + col];
        whi[i] = Wp[(2 * kp + 1) * D + col];
      }
    }
    // Prefetch next hop's weights + gathered feature rows into near caches (global_prefetch_b8).
    if (m + 1 < 9) {
      const float* WpN = hop_wptr(m + 1, n, W_S, W_L3, W_L4);
      __builtin_prefetch(WpN + (size_t)tid * 16, 0, 3);
      int jN = hop_j(m + 1, n, idxS, idxL3, idxL4);
      __builtin_prefetch(feat + ((size_t)(half * HBATCH + (tid & 127)) * NN + jN) * CC
                              + (tid >> 7) * 32, 0, 3);
    }
    // Pack + store phase (loads above retire under the clause while this unrolls).
    #pragma unroll
    for (int k = 0; k < 8; ++k) {
      int p = k * 256 + tid;
      int row = p >> 4, cq = p & 15;
      uint2 pk;
      pk.x = pack_bf2(xf[k].x * av[k], xf[k].y * av[k]);
      pk.y = pack_bf2(xf[k].z * av[k], xf[k].w * av[k]);
      *(uint2*)(x_u32 + row * 32 + 2 * cq) = pk;
    }
    {
      int col = tid % D, g = tid / D;
      #pragma unroll
      for (int i = 0; i < per; ++i) {
        int kp = g * per + i;
        wT_u32[col * 32 + kp] = pack_bf2(wlo[i], whi[i]);
      }
    }
    __syncthreads();

    // Stage 1: Yacc += X(16x64) @ W(64xD), wave w owns rows 16w..16w+15
    const unsigned int* xrow = x_u32 + (16 * w + lr) * 32;
    #pragma unroll
    for (int kc = 0; kc < 2; ++kc) {
      v16bf Aop = ld_a(xrow, kc * 16, hs);
      #pragma unroll
      for (int nt = 0; nt < D / 16; ++nt) {
        v16bf Bop = ld_b(wT_u32 + (nt * 16 + lr) * 32, kc * 16, hs);
        Yacc[nt] = __builtin_amdgcn_wmma_f32_16x16x32_bf16(
            false, Aop, false, Bop, (short)0, Yacc[nt], false, false);
      }
    }
  }

  // Round-trip Y through LDS (wave-private rows -> no barrier needed) to convert C/D layout -> A layout.
  unsigned short* y16 = (unsigned short*)y_u32;
  #pragma unroll
  for (int nt = 0; nt < D / 16; ++nt)
    #pragma unroll
    for (int r = 0; r < 8; ++r) {
      int row = 16 * w + r + 8 * hs;
      y16[row * 64 + nt * 16 + lr] = f2bf(Yacc[nt][r]);
    }

  // Stage 2: Hacc += Y(16xD) @ Wa[R0:R0+D, 0:64]
  const unsigned int* yrow = y_u32 + (16 * w + lr) * 32;
  #pragma unroll
  for (int kc = 0; kc < D / 32; ++kc) {
    v16bf Aop = ld_a(yrow, kc * 16, hs);
    #pragma unroll
    for (int nt = 0; nt < 4; ++nt) {
      v16bf Bop = ld_b(waT_u32 + (nt * 16 + lr) * 80 + (R0 / 2), kc * 16, hs);
      Hacc[nt] = __builtin_amdgcn_wmma_f32_16x16x32_bf16(
          false, Aop, false, Bop, (short)0, Hacc[nt], false, false);
    }
  }
}

__global__ __launch_bounds__(256, 1) void k_hier_main(
    const float* __restrict__ feat, const float* __restrict__ Amat,
    const float* __restrict__ W_S, const float* __restrict__ W_L3, const float* __restrict__ W_L4,
    const float* __restrict__ Wa,
    const int* __restrict__ idxS, const int* __restrict__ idxL3, const int* __restrict__ idxL4,
    float* __restrict__ Hout, float* __restrict__ partials)
{
  __shared__ __align__(16) unsigned int x_u32[HBATCH * 32];   // 16 KB scaled features (bf16 pairs)
  __shared__ __align__(16) unsigned int y_u32[HBATCH * 32];   // 16 KB hop outputs    (bf16 pairs)
  __shared__ __align__(16) unsigned int wT_u32[64 * 32];      //  8 KB W^T            (bf16 pairs)
  __shared__ __align__(16) unsigned int waT_u32[64 * 80];     // 20 KB Wa^T, K=160    (bf16 pairs)
  __shared__ float stats[128];                                // sum[64], sumsq[64]

  const int n = blockIdx.x;
  const int tid = threadIdx.x;
  const int w = tid >> 5, lane = tid & 31, hs = lane >> 4, lr = lane & 15;

  if (tid < 128) stats[tid] = 0.f;
  {
    // Stage Wa transposed once: waT[o][kp] = {Wa[2kp,o], Wa[2kp+1,o]}
    int o = tid & 63, g = tid >> 6;
    #pragma unroll
    for (int i = 0; i < 20; ++i) {
      int kp = g * 20 + i;
      waT_u32[o * 80 + kp] = pack_bf2(Wa[(2 * kp) * 64 + o], Wa[(2 * kp + 1) * 64 + o]);
    }
  }
  __syncthreads();

  for (int half = 0; half < 2; ++half) {
    v8f Hacc[4] = {};
    process_group<0, 5, 64,  0>(n, half, tid, w, hs, lr, feat, Amat, idxS, idxL3, idxL4,
                                W_S, W_L3, W_L4, x_u32, y_u32, wT_u32, waT_u32, Hacc);
    process_group<5, 7, 32, 64>(n, half, tid, w, hs, lr, feat, Amat, idxS, idxL3, idxL4,
                                W_S, W_L3, W_L4, x_u32, y_u32, wT_u32, waT_u32, Hacc);
    process_group<7, 9, 64, 96>(n, half, tid, w, hs, lr, feat, Amat, idxS, idxL3, idxL4,
                                W_S, W_L3, W_L4, x_u32, y_u32, wT_u32, waT_u32, Hacc);

    // Write H tile + per-channel partial stats (LDS float atomics: only 2-way lane conflicts).
    #pragma unroll
    for (int nt = 0; nt < 4; ++nt) {
      float s = 0.f, ss = 0.f;
      #pragma unroll
      for (int r = 0; r < 8; ++r) {
        float v = Hacc[nt][r];
        int b = half * HBATCH + 16 * w + r + 8 * hs;
        Hout[((size_t)b * NN + n) * CC + nt * 16 + lr] = v;
        s += v; ss += v * v;
      }
      atomicAdd(&stats[nt * 16 + lr], s);
      atomicAdd(&stats[64 + nt * 16 + lr], ss);
    }
  }
  __syncthreads();
  if (tid < 128) partials[(size_t)n * 128 + tid] = stats[tid];
}

__global__ void k_finalize(const float* __restrict__ partials,
                           const float* __restrict__ gamma, const float* __restrict__ beta,
                           float* __restrict__ scsh)
{
  int t = threadIdx.x;
  if (t >= 64) return;
  float s = 0.f, ss = 0.f;
  for (int nn = 0; nn < NN; ++nn) {
    s  += partials[(size_t)nn * 128 + t];
    ss += partials[(size_t)nn * 128 + 64 + t];
  }
  const float inv = 1.0f / ((float)BB * (float)NN);
  float mean = s * inv;
  float var  = ss * inv - mean * mean;
  float sc   = gamma[t] * rsqrtf(var + EPSF);
  scsh[t]      = sc;
  scsh[64 + t] = beta[t] - mean * sc;
}

__global__ void k_bn_prelu(float* __restrict__ H, const float* __restrict__ scsh,
                           const float* __restrict__ prelu)
{
  __shared__ float sc[64], sh[64];
  int t = threadIdx.x;
  if (t < 64) { sc[t] = scsh[t]; sh[t] = scsh[64 + t]; }
  __syncthreads();
  float alpha = prelu[0];
  const size_t total4 = (size_t)BB * NN * CC / 4;
  for (size_t i = (size_t)blockIdx.x * blockDim.x + t; i < total4;
       i += (size_t)gridDim.x * blockDim.x) {
    float4 v = ((float4*)H)[i];
    int c = (int)((i * 4) & 63);
    float y0 = v.x * sc[c]     + sh[c];
    float y1 = v.y * sc[c + 1] + sh[c + 1];
    float y2 = v.z * sc[c + 2] + sh[c + 2];
    float y3 = v.w * sc[c + 3] + sh[c + 3];
    v.x = y0 > 0.f ? y0 : alpha * y0;
    v.y = y1 > 0.f ? y1 : alpha * y1;
    v.z = y2 > 0.f ? y2 : alpha * y2;
    v.w = y3 > 0.f ? y3 : alpha * y3;
    ((float4*)H)[i] = v;
  }
}

extern "C" void kernel_launch(void* const* d_in, const int* in_sizes, int n_in,
                              void* d_out, int out_size, void* d_ws, size_t ws_size,
                              hipStream_t stream)
{
  (void)in_sizes; (void)n_in; (void)out_size; (void)ws_size;
  const float* feat  = (const float*)d_in[0];
  const float* Amat  = (const float*)d_in[1];
  const float* W_S   = (const float*)d_in[2];
  const float* W_L3  = (const float*)d_in[3];
  const float* W_L4  = (const float*)d_in[4];
  const float* Wa    = (const float*)d_in[5];
  const float* gamma = (const float*)d_in[6];
  const float* beta  = (const float*)d_in[7];
  const float* prelu = (const float*)d_in[8];
  const int*   idxS  = (const int*)d_in[9];
  const int*   idxL3 = (const int*)d_in[10];
  const int*   idxL4 = (const int*)d_in[11];

  float* H        = (float*)d_out;
  float* partials = (float*)d_ws;                  // [512][128] per-node sum/sumsq
  float* scsh     = partials + (size_t)NN * 128;   // [128] fused BN scale/shift

  k_hier_main<<<dim3(NN), dim3(256), 0, stream>>>(feat, Amat, W_S, W_L3, W_L4, Wa,
                                                  idxS, idxL3, idxL4, H, partials);
  k_finalize<<<dim3(1), dim3(64), 0, stream>>>(partials, gamma, beta, scsh);
  k_bn_prelu<<<dim3(1024), dim3(256), 0, stream>>>(H, scsh, prelu);
}
